// Bah_Attn_51745765982216
// MI455X (gfx1250) — matrix operand order repaired and checked by reading, not compiled, not run on hardware
//
#include <hip/hip_runtime.h>

// Bahdanau attention, fused for MI455X (gfx1250, wave32, WMMA bf16).
// Pipeline: zero -> lin1 (q = query@W1+b1) -> fused lin2+score (WMMA bf16)
//           -> softmax -> context GEMV.

typedef __attribute__((ext_vector_type(16))) __bf16 v16bf;
typedef __attribute__((ext_vector_type(2)))  __bf16 v2bf;
typedef __attribute__((ext_vector_type(8)))  float  v8f;

#define B_SZ 32
#define S_SZ 2048
#define D_SZ 2048   // HIDDEN * DIRECTION (GEMM K)
#define H_SZ 1024   // HIDDEN (GEMM N)

// fp32 -> bf16 round-to-nearest-even (bit trick fallback)
__device__ __forceinline__ unsigned short f2bf(float f) {
    unsigned int u = __float_as_uint(f);
    u += 0x7fffu + ((u >> 16) & 1u);
    return (unsigned short)(u >> 16);
}

union V2U { v2bf v; unsigned short u[2]; unsigned int i; };

// packed 2xf32 -> 2xbf16: v_cvt_pk_bf16_f32 on gfx1250 if available
__device__ __forceinline__ V2U pk2bf(float x, float y) {
    V2U r;
#if __has_builtin(__builtin_amdgcn_cvt_pk_bf16_f32)
    r.v = __builtin_amdgcn_cvt_pk_bf16_f32(x, y);
#else
    r.u[0] = f2bf(x);
    r.u[1] = f2bf(y);
#endif
    return r;
}

// gfx1250 hardware tanh (v_tanh_f32) if the builtin exists; else libm expansion.
__device__ __forceinline__ float fast_tanh(float x) {
#if __has_builtin(__builtin_amdgcn_tanhf)
    return __builtin_amdgcn_tanhf(x);
#else
    return tanhf(x);
#endif
}

union V16U { v16bf v; v2bf h[8]; unsigned short u[16]; uint4 q[2]; };

// ---------------------------------------------------------------- zero scratch
__global__ void zero_f32(float* __restrict__ p, int n) {
    int i = blockIdx.x * blockDim.x + threadIdx.x;
    if (i < n) p[i] = 0.f;
}

// ---------------------------------------------------------------- q = query@W1 + b1
// 67 MFLOP total; plain VALU fp32 is fine. grid (32, 4), block 256.
__global__ __launch_bounds__(256)
void lin1_kernel(const float* __restrict__ query, const float* __restrict__ W1,
                 const float* __restrict__ b1, float* __restrict__ qbuf) {
    const int b = blockIdx.x;
    const int n = blockIdx.y * 256 + threadIdx.x;
    const float* qr = query + b * 1024;
    float acc = b1[n];
    #pragma unroll 4
    for (int k = 0; k < 1024; ++k)
        acc = fmaf(qr[k], W1[(size_t)k * H_SZ + n], acc);
    qbuf[b * H_SZ + n] = acc;
}

// ---------------------------------------------------------------- fused lin2 + score
// Block tile: M=64 rows of (b,s), N=256 cols of h, full K=2048.
// 8 waves: wave = (m_tile[0..3], n_group[0..1]); each wave owns 8 16x16 C tiles.
// W2 slab (32 K x 256 N) staged to LDS as bf16, transposed, stride 40 (80B) to
// keep ds_load_b128 B-fragment reads bank-conflict free. A (key) fragments are
// converted fp32->bf16 (v_cvt_pk_bf16_f32) in registers straight from global.
// B fragments are double-buffered so ds_load_b128 overlaps v_wmma.
// Epilogue fuses sum_h v[h]*tanh(q[b,h]+b2[h]+lk) with wave32 butterfly reduce
// + global_atomic_add_f32 across the 4 N-blocks.
__global__ __launch_bounds__(256)
void lin2_score_kernel(const float* __restrict__ key,
                       const float* __restrict__ W2,
                       const float* __restrict__ b2,
                       const float* __restrict__ vvec,
                       const float* __restrict__ qbuf,
                       float* __restrict__ score_raw) {
    __shared__ unsigned short bslab[256 * 40];   // 20 KB

    const int tid  = threadIdx.x;
    const int lane = tid & 31;
    const int wave = tid >> 5;
    const int m0   = blockIdx.x * 64;            // 1024 M-blocks
    const int n0   = blockIdx.y * 256;           // 4 N-blocks
    const int b    = m0 >> 11;                   // all 64 rows share batch (2048 % 64 == 0)
    const int mt   = wave >> 1;                  // m-tile 0..3
    const int ng   = wave & 1;                   // n-group 0..1 (128 cols each)
    const int l16  = lane & 15;
    const int hi   = lane >> 4;                  // half-wave select

    const float* akrow = key + (size_t)(m0 + mt * 16 + l16) * D_SZ;
    // B fragment base for this lane: tiles are 80 uint4 (1280 B) apart
    const uint4* bbase = (const uint4*)(&bslab[(ng * 128 + l16) * 40 + hi * 16]);

    v8f acc[8];
    const v8f vzero = {0.f, 0.f, 0.f, 0.f, 0.f, 0.f, 0.f, 0.f};
    #pragma unroll
    for (int j = 0; j < 8; ++j) acc[j] = vzero;

    for (int k0 = 0; k0 < D_SZ; k0 += 32) {
        // --- prefetch next W2 slab (global_prefetch_b8, near-cache locality)
        if (k0 + 32 < D_SZ) {
            const float* pf = W2 + (size_t)(k0 + 32 + (tid >> 3)) * H_SZ
                                 + n0 + (tid & 7) * 32;
            __builtin_prefetch(pf, 0, 3);
        }

        // --- cooperative stage of W2[k0:k0+32, n0:n0+256] -> LDS (bf16, [n][k])
        #pragma unroll
        for (int i = 0; i < 8; ++i) {
            int fi = tid + i * 256;              // 0..2047 float4 index
            int k  = fi >> 6;                    // 0..31
            int n  = (fi & 63) * 4;              // 0..252
            float4 w = *(const float4*)(W2 + (size_t)(k0 + k) * H_SZ + n0 + n);
            V2U p0 = pk2bf(w.x, w.y);
            V2U p1 = pk2bf(w.z, w.w);
            bslab[(n + 0) * 40 + k] = (unsigned short)(p0.i & 0xffffu);
            bslab[(n + 1) * 40 + k] = (unsigned short)(p0.i >> 16);   // ds_store_b16_d16_hi
            bslab[(n + 2) * 40 + k] = (unsigned short)(p1.i & 0xffffu);
            bslab[(n + 3) * 40 + k] = (unsigned short)(p1.i >> 16);
        }

        // --- A fragment: 16x32 bf16 (contiguous k = 16*hi + s, same order as B), packed cvt in regs
        const float4* ap = (const float4*)(akrow + k0 + hi * 16);
        float4 a0 = ap[0], a1 = ap[1], a2 = ap[2], a3 = ap[3];
        V16U af;
        af.h[0] = pk2bf(a0.x, a0.y).v;
        af.h[1] = pk2bf(a0.z, a0.w).v;
        af.h[2] = pk2bf(a1.x, a1.y).v;
        af.h[3] = pk2bf(a1.z, a1.w).v;
        af.h[4] = pk2bf(a2.x, a2.y).v;
        af.h[5] = pk2bf(a2.z, a2.w).v;
        af.h[6] = pk2bf(a3.x, a3.y).v;
        af.h[7] = pk2bf(a3.z, a3.w).v;

        __syncthreads();

        // --- 8 WMMAs per wave per K-step; B fragments double-buffered so the
        //     ds_load_b128 pair for tile j+1 is in flight during wmma of tile j
        V16U bf0, bf1;
        bf0.q[0] = bbase[0];
        bf0.q[1] = bbase[1];
        #pragma unroll
        for (int j = 0; j < 8; ++j) {
            V16U& cur = (j & 1) ? bf1 : bf0;
            V16U& nxt = (j & 1) ? bf0 : bf1;
            if (j + 1 < 8) {
                nxt.q[0] = bbase[(j + 1) * 80];
                nxt.q[1] = bbase[(j + 1) * 80 + 1];
            }
            acc[j] = __builtin_amdgcn_wmma_f32_16x16x32_bf16(
                         false, af.v, false, cur.v, (short)0, acc[j], false, false);
        }
        __syncthreads();
    }

    // --- fused score epilogue: p[row] += v[n] * tanh(q[b,n] + b2[n] + lk)
    float p[8] = {0.f, 0.f, 0.f, 0.f, 0.f, 0.f, 0.f, 0.f};
    #pragma unroll
    for (int j = 0; j < 8; ++j) {
        int ncol = n0 + ng * 128 + j * 16 + l16;
        float t  = qbuf[b * H_SZ + ncol] + b2[ncol];
        float vv = vvec[ncol];
        #pragma unroll
        for (int i = 0; i < 8; ++i)
            p[i] += fast_tanh(t + acc[j][i]) * vv;
    }
    // butterfly over the 16 lanes sharing each row group (wave32)
    #pragma unroll
    for (int m = 1; m < 16; m <<= 1) {
        #pragma unroll
        for (int i = 0; i < 8; ++i) p[i] += __shfl_xor(p[i], m, 32);
    }
    if (l16 == 0) {
        int row = m0 + mt * 16 + hi * 8;         // C layout: VGPR i -> row i + 8*hi
        #pragma unroll
        for (int i = 0; i < 8; ++i)
            atomicAdd(&score_raw[row + i], p[i]);
    }
}

// ---------------------------------------------------------------- softmax over S per batch
__global__ __launch_bounds__(256)
void softmax_kernel(const float* __restrict__ raw, const float* __restrict__ bvp,
                    float* __restrict__ outscore) {
    __shared__ float redmax[8];
    __shared__ float redsum[8];
    const int b = blockIdx.x, tid = threadIdx.x;
    const int lane = tid & 31, wave = tid >> 5;
    const float bv = bvp[0];

    float x[8];
    float mx = -3.0e38f;
    #pragma unroll
    for (int i = 0; i < 8; ++i) {
        x[i] = raw[b * S_SZ + tid + i * 256] + bv;
        mx = fmaxf(mx, x[i]);
    }
    #pragma unroll
    for (int m = 16; m >= 1; m >>= 1) mx = fmaxf(mx, __shfl_xor(mx, m, 32));
    if (lane == 0) redmax[wave] = mx;
    __syncthreads();
    mx = redmax[0];
    #pragma unroll
    for (int w = 1; w < 8; ++w) mx = fmaxf(mx, redmax[w]);

    float s = 0.f;
    #pragma unroll
    for (int i = 0; i < 8; ++i) { x[i] = __expf(x[i] - mx); s += x[i]; }
    #pragma unroll
    for (int m = 16; m >= 1; m >>= 1) s += __shfl_xor(s, m, 32);
    if (lane == 0) redsum[wave] = s;
    __syncthreads();
    s = 0.f;
    #pragma unroll
    for (int w = 0; w < 8; ++w) s += redsum[w];

    const float inv = 1.f / s;
    #pragma unroll
    for (int i = 0; i < 8; ++i)
        outscore[b * S_SZ + tid + i * 256] = x[i] * inv;
}

// ---------------------------------------------------------------- context = score @ key
// Memory-bound (512 MB key pass): float2/thread, coalesced b64 loads over d,
// score broadcast via SALU. grid: 32 b * 4 chunks of 512 floats.
__global__ __launch_bounds__(256)
void context_kernel(const float* __restrict__ score, const float* __restrict__ key,
                    float* __restrict__ ctx) {
    const int b = blockIdx.x >> 2;
    const int d = ((blockIdx.x & 3) << 9) + threadIdx.x * 2;
    const float2* kb = (const float2*)(key + (size_t)b * S_SZ * D_SZ + d);
    const float* sb = score + b * S_SZ;
    float ax = 0.f, ay = 0.f;
    for (int s = 0; s < S_SZ; s += 4) {
        float s0 = sb[s], s1 = sb[s + 1], s2 = sb[s + 2], s3 = sb[s + 3];
        float2 k0 = kb[(size_t)(s + 0) * (D_SZ / 2)];
        float2 k1 = kb[(size_t)(s + 1) * (D_SZ / 2)];
        float2 k2 = kb[(size_t)(s + 2) * (D_SZ / 2)];
        float2 k3 = kb[(size_t)(s + 3) * (D_SZ / 2)];
        ax = fmaf(s0, k0.x, ax); ay = fmaf(s0, k0.y, ay);
        ax = fmaf(s1, k1.x, ax); ay = fmaf(s1, k1.y, ay);
        ax = fmaf(s2, k2.x, ax); ay = fmaf(s2, k2.y, ay);
        ax = fmaf(s3, k3.x, ax); ay = fmaf(s3, k3.y, ay);
    }
    ctx[b * D_SZ + d]     = ax;
    ctx[b * D_SZ + d + 1] = ay;
}

// ---------------------------------------------------------------- launcher
extern "C" void kernel_launch(void* const* d_in, const int* in_sizes, int n_in,
                              void* d_out, int out_size, void* d_ws, size_t ws_size,
                              hipStream_t stream) {
    const float* query = (const float*)d_in[0];   // [32, 1024]
    const float* key   = (const float*)d_in[1];   // [32, 2048, 2048]
    const float* W1    = (const float*)d_in[2];   // [1024, 1024]
    const float* b1    = (const float*)d_in[3];   // [1024]
    const float* W2    = (const float*)d_in[4];   // [2048, 1024]
    const float* b2    = (const float*)d_in[5];   // [1024]
    const float* vv    = (const float*)d_in[6];   // [1024]
    const float* bv    = (const float*)d_in[7];   // scalar

    float* out   = (float*)d_out;
    float* ctx   = out;                       // [32, 2048]
    float* score = out + B_SZ * D_SZ;         // [32, 2048]

    float* qbuf = (float*)d_ws;               // [32, 1024]  = 128 KB
    float* raw  = qbuf + B_SZ * H_SZ;         // [32, 2048]  = 256 KB

    zero_f32<<<(B_SZ * S_SZ + 255) / 256, 256, 0, stream>>>(raw, B_SZ * S_SZ);
    lin1_kernel<<<dim3(B_SZ, 4), 256, 0, stream>>>(query, W1, b1, qbuf);
    lin2_score_kernel<<<dim3((B_SZ * S_SZ) / 64, 4), 256, 0, stream>>>(
        key, W2, b2, vv, qbuf, raw);
    softmax_kernel<<<B_SZ, 256, 0, stream>>>(raw, bv, score);
    context_kernel<<<B_SZ * 8 / 2, 256, 0, stream>>>(score, key, ctx);
}
